// TransformerBlock_72730976190493
// MI455X (gfx1250) — matrix operand order repaired and checked by reading, not compiled, not run on hardware
//
#include <hip/hip_runtime.h>
#include <hip/hip_bf16.h>
#include <stdint.h>

typedef __attribute__((ext_vector_type(16))) _Float16 v16h;
typedef __attribute__((ext_vector_type(8)))  _Float16 v8h;
typedef __attribute__((ext_vector_type(8)))  float    v8f;

#define TB_B    2
#define TB_T    2048
#define TB_C    1024
#define TB_H    16
#define TB_D    64
#define TB_E    4
#define TB_HID  2730
#define TB_HIDP 2816   /* padded to multiple of 128 */
#define TB_M    4096   /* B*T */
#define TB_3C   3072

// ---------------------------------------------------------------------------
// WMMA helpers (CDNA5 gfx1250, wave32)
// ---------------------------------------------------------------------------
__device__ __forceinline__ v8f wmma_f16(v16h a, v16h b, v8f c) {
  // v_wmma_f32_16x16x32_f16
  return __builtin_amdgcn_wmma_f32_16x16x32_f16(false, a, false, b, (short)0, c,
                                                false, false);
}

// A-fragment 16x32 f16 (ISA 7.12.2): lane m = lane%16, half = lane/16.
// element i -> k = (i<8) ? 8*half+i : 16+8*half+(i-8). Two 16B loads.
__device__ __forceinline__ v16h ld_fragA(const _Float16* __restrict__ p, int half) {
  v8h lo = *(const v8h*)(p + 8 * half);
  v8h hi = *(const v8h*)(p + 16 + 8 * half);
  v16h f;
#pragma unroll
  for (int i = 0; i < 8; ++i) { f[i] = lo[i]; f[i + 8] = hi[i]; }
  return f;
}

// B-fragment 32x16 f16 read from transposed weights BT[n][k]:
// lane n = lane%16, half = lane/16; element i -> k = (i<8) ? 8*half+i : 16+8*half+(i-8).
__device__ __forceinline__ v16h ld_fragB(const _Float16* __restrict__ p, int half) {
  v8h lo = *(const v8h*)(p + 8 * half);
  v8h hi = *(const v8h*)(p + 16 + 8 * half);
  v16h f;
#pragma unroll
  for (int i = 0; i < 8; ++i) { f[i] = lo[i]; f[i + 8] = hi[i]; }
  return f;
}

// CDNA5 async global->LDS copy (ASYNCcnt-tracked), 16B per active lane.
// Low 32 bits of a generic LDS pointer are the LDS byte address (aperture rule).
__device__ __forceinline__ void async_copy_b128(void* lds, const _Float16* gptr) {
  uint32_t loff = (uint32_t)(uintptr_t)lds;
  asm volatile("global_load_async_to_lds_b128 %0, %1, off"
               :: "v"(loff), "v"(gptr) : "memory");
}
__device__ __forceinline__ void wait_async0() {
  asm volatile("s_wait_asynccnt 0x0" ::: "memory");
}

// ---------------------------------------------------------------------------
// fp32 [K][N] -> f16 transposed [OR][OC] with zero padding (OR>=N, OC>=K)
// ---------------------------------------------------------------------------
__global__ void tb_transpose_f16(const float* __restrict__ in, int K, int N,
                                 _Float16* __restrict__ out, int OR, int OC) {
  __shared__ float tile[32][33];
  int c0 = blockIdx.x * 32;  // out col block == in row block
  int r0 = blockIdx.y * 32;  // out row block == in col block
  int tx = threadIdx.x, ty = threadIdx.y;
#pragma unroll
  for (int i = 0; i < 32; i += 8) {
    int k = c0 + ty + i, n = r0 + tx;
    tile[ty + i][tx] = (k < K && n < N) ? in[(size_t)k * N + n] : 0.f;
  }
  __syncthreads();
#pragma unroll
  for (int i = 0; i < 32; i += 8) {
    int r = r0 + ty + i, c = c0 + tx;
    if (r < OR && c < OC) out[(size_t)r * OC + c] = (_Float16)tile[tx][ty + i];
  }
}

// ---------------------------------------------------------------------------
// RMSNorm: fp32 in -> f16 out, one block per token
// ---------------------------------------------------------------------------
__global__ void tb_rmsnorm(const float* __restrict__ x, const float* __restrict__ g,
                           _Float16* __restrict__ out) {
  int t = blockIdx.x;
  const float* xr = x + (size_t)t * TB_C;
  __shared__ float red[256];
  float s = 0.f;
  for (int c = threadIdx.x; c < TB_C; c += 256) { float v = xr[c]; s += v * v; }
  red[threadIdx.x] = s;
  __syncthreads();
  for (int st = 128; st > 0; st >>= 1) {
    if (threadIdx.x < st) red[threadIdx.x] += red[threadIdx.x + st];
    __syncthreads();
  }
  float scale = rsqrtf(red[0] * (1.f / TB_C) + 1e-6f);
  for (int c = threadIdx.x; c < TB_C; c += 256)
    out[(size_t)t * TB_C + c] = (_Float16)(xr[c] * scale * g[c]);
}

// ---------------------------------------------------------------------------
// WMMA GEMM with async double-buffered LDS staging.
// C[M,N] = A[M,K](f16) * BT[N,K](f16)^T (+bias) + epilogue.
// 256 thr = 8 waves; block tile 128x128; wave tile 32x64 (8 accumulators).
// K consumed in 32-wide slices: As/Bs [128 rows][32 k] (row stride 40 halves
// to dodge bank conflicts while keeping 16B alignment), double buffered.
// EPI: 0=f16 store, 1=fp32 residual store, 2=SiLU f16, 3=mul-by-gate f16,
//      4=MoE weighted accumulate into fp32
// ---------------------------------------------------------------------------
#define LDW 40
template <int EPI>
__launch_bounds__(256)
__global__ void tb_gemm(const _Float16* __restrict__ A, const _Float16* __restrict__ BT,
                        const float* __restrict__ bias, int M, int N, int K, int Nvalid,
                        _Float16* __restrict__ outH, float* __restrict__ outF,
                        const float* __restrict__ resid, const _Float16* __restrict__ mulH,
                        const float* __restrict__ wts, int eidx,
                        float* __restrict__ accF) {
  __shared__ __align__(16) _Float16 As[2][128 * LDW];
  __shared__ __align__(16) _Float16 Bs[2][128 * LDW];

  int tid = threadIdx.x;
  int wave = tid >> 5, lane = tid & 31;
  int lr = lane & 15, half = lane >> 4;
  int wm = wave & 3, wn = wave >> 2;   // 4 x 2 waves -> wave tile 32(M) x 64(N)
  int tM = blockIdx.x * 128;
  int tN = blockIdx.y * 128;

  // Each thread async-copies 2x16B for A and 2x16B for B per k-slice:
  // 128 rows x 4 16B-parts = 512 segments, 256 threads.
  int r0 = tid >> 2, p0 = (tid & 3) * 8;
  int r1 = (tid + 256) >> 2, p1 = ((tid + 256) & 3) * 8;

  v8f acc[2][4] = {};
  int nk = K >> 5;

  // prologue: stage slice 0
  {
    async_copy_b128(&As[0][r0 * LDW + p0], A + (size_t)(tM + r0) * K + p0);
    async_copy_b128(&As[0][r1 * LDW + p1], A + (size_t)(tM + r1) * K + p1);
    async_copy_b128(&Bs[0][r0 * LDW + p0], BT + (size_t)(tN + r0) * K + p0);
    async_copy_b128(&Bs[0][r1 * LDW + p1], BT + (size_t)(tN + r1) * K + p1);
  }

  for (int kc = 0; kc < nk; ++kc) {
    wait_async0();
    __syncthreads();
    if (kc + 1 < nk) {  // stage next slice into the other buffer (overlapped)
      int k0 = (kc + 1) << 5, nb = (kc + 1) & 1;
      async_copy_b128(&As[nb][r0 * LDW + p0], A + (size_t)(tM + r0) * K + k0 + p0);
      async_copy_b128(&As[nb][r1 * LDW + p1], A + (size_t)(tM + r1) * K + k0 + p1);
      async_copy_b128(&Bs[nb][r0 * LDW + p0], BT + (size_t)(tN + r0) * K + k0 + p0);
      async_copy_b128(&Bs[nb][r1 * LDW + p1], BT + (size_t)(tN + r1) * K + k0 + p1);
    }
    const _Float16* Ab = &As[kc & 1][0];
    const _Float16* Bb = &Bs[kc & 1][0];
    v16h a0 = ld_fragA(Ab + (wm * 32 + lr) * LDW, half);
    v16h a1 = ld_fragA(Ab + (wm * 32 + 16 + lr) * LDW, half);
#pragma unroll
    for (int ni = 0; ni < 4; ++ni) {
      v16h b = ld_fragB(Bb + (wn * 64 + ni * 16 + lr) * LDW, half);
      acc[0][ni] = wmma_f16(a0, b, acc[0][ni]);
      acc[1][ni] = wmma_f16(a1, b, acc[1][ni]);
    }
    __syncthreads();
  }

#pragma unroll
  for (int mi = 0; mi < 2; ++mi)
#pragma unroll
    for (int ni = 0; ni < 4; ++ni)
#pragma unroll
      for (int r = 0; r < 8; ++r) {
        int m = tM + wm * 32 + mi * 16 + r + 8 * half;  // C/D: row = r+8*(lane/16)
        int n = tN + wn * 64 + ni * 16 + lr;            // col = lane%16
        float v = acc[mi][ni][r];
        if (n < Nvalid) v += bias[n];
        if constexpr (EPI == 0) {
          outH[(size_t)m * N + n] = (_Float16)v;
        } else if constexpr (EPI == 1) {
          outF[(size_t)m * N + n] = resid[(size_t)m * N + n] + v;
        } else if constexpr (EPI == 2) {
          float sg = 1.f / (1.f + __expf(-v));
          outH[(size_t)m * N + n] = (_Float16)(v * sg);
        } else if constexpr (EPI == 3) {
          float gv = (float)mulH[(size_t)m * N + n];
          outH[(size_t)m * N + n] = (_Float16)(v * gv);
        } else {
          float w = wts[(size_t)m * TB_E + eidx];
          float prev = (eidx == 0) ? 0.f : accF[(size_t)m * N + n];
          accF[(size_t)m * N + n] = prev + w * v;
        }
      }
}

// ---------------------------------------------------------------------------
// RoPE + reshape: qkv f16 [M][3C] -> Q[BH][T][D], K[BH][T][D], VT[BH][D][T]
// ---------------------------------------------------------------------------
__global__ void tb_rope(const _Float16* __restrict__ qkv, _Float16* __restrict__ Q,
                        _Float16* __restrict__ Ko, _Float16* __restrict__ VT) {
  int t = blockIdx.x;
  int b = t >> 11, tt = t & 2047;
  const _Float16* row = qkv + (size_t)t * TB_3C;
  for (int idx = threadIdx.x; idx < TB_3C; idx += 256) {
    int s = idx >> 10, rem = idx & 1023;
    int h = rem >> 6, d = rem & 63;
    int bh = b * TB_H + h;
    if (s == 2) {
      VT[((size_t)bh * TB_D + d) * TB_T + tt] = row[idx];
    } else {
      int j = d >> 1;
      float ang = (float)tt * __powf(10000.f, -(float)(2 * j) * (1.f / TB_D));
      float sn, cs;
      __sincosf(ang, &sn, &cs);
      float xe = (float)row[(s << 10) + (h << 6) + 2 * j];
      float xo = (float)row[(s << 10) + (h << 6) + 2 * j + 1];
      float v = (d & 1) ? (xe * sn + xo * cs) : (xe * cs - xo * sn);
      _Float16 hv = (_Float16)v;
      if (s == 0) Q[((size_t)bh * TB_T + tt) * TB_D + d] = hv;
      else        Ko[((size_t)bh * TB_T + tt) * TB_D + d] = hv;
    }
  }
}

// ---------------------------------------------------------------------------
// Flash attention, causal. 128 thr = 4 waves; wave owns 16 query rows.
// Scores & PV through v_wmma_f32_16x16x32_f16; online softmax w/ wave32 shfl.
// ---------------------------------------------------------------------------
__launch_bounds__(128)
__global__ void tb_attn(const _Float16* __restrict__ Q, const _Float16* __restrict__ Kc,
                        const _Float16* __restrict__ VT, _Float16* __restrict__ O) {
  int bh = blockIdx.y;
  int b = bh >> 4, h = bh & 15;
  int wave = threadIdx.x >> 5, lane = threadIdx.x & 31;
  int lr = lane & 15, half = lane >> 4;
  int q0 = blockIdx.x * 64 + wave * 16;

  const _Float16* q  = Q  + (size_t)bh * TB_T * TB_D;
  const _Float16* kk = Kc + (size_t)bh * TB_T * TB_D;
  const _Float16* vt = VT + (size_t)bh * TB_D * TB_T;

  __shared__ __align__(16) _Float16 pb[4][16 * 32];
  _Float16* myP = pb[wave];

  v16h qf0 = ld_fragA(q + (size_t)(q0 + lr) * TB_D, half);
  v16h qf1 = ld_fragA(q + (size_t)(q0 + lr) * TB_D + 32, half);

  v8f o_acc[4] = {};
  float runm[8], runs[8], alpha[8];
#pragma unroll
  for (int r = 0; r < 8; ++r) { runm[r] = -3e38f; runs[r] = 0.f; }

  int kbEnd = (q0 + 15) >> 5;
  for (int kb = 0; kb <= kbEnd; ++kb) {
    int kbase = kb * 32;
    v8f s0 = {}, s1 = {};
    {
      v16h kf0 = ld_fragB(kk + (size_t)(kbase + lr) * TB_D, half);
      v16h kf1 = ld_fragB(kk + (size_t)(kbase + lr) * TB_D + 32, half);
      s0 = wmma_f16(qf0, kf0, s0);
      s0 = wmma_f16(qf1, kf1, s0);
      v16h kg0 = ld_fragB(kk + (size_t)(kbase + 16 + lr) * TB_D, half);
      v16h kg1 = ld_fragB(kk + (size_t)(kbase + 16 + lr) * TB_D + 32, half);
      s1 = wmma_f16(qf0, kg0, s1);
      s1 = wmma_f16(qf1, kg1, s1);
    }
    // scale + causal mask (C layout: row = r+8*half, col = lane%16)
#pragma unroll
    for (int r = 0; r < 8; ++r) {
      int qg = q0 + r + 8 * half;
      float a = s0[r] * 0.125f;
      float c = s1[r] * 0.125f;
      if (kbase + lr > qg) a = -3e38f;
      if (kbase + 16 + lr > qg) c = -3e38f;
      s0[r] = a; s1[r] = c;
    }
    // online softmax: reduce over the 16 lanes of each half-wave
#pragma unroll
    for (int r = 0; r < 8; ++r) {
      float mx = fmaxf(s0[r], s1[r]);
      mx = fmaxf(mx, __shfl_xor(mx, 1));
      mx = fmaxf(mx, __shfl_xor(mx, 2));
      mx = fmaxf(mx, __shfl_xor(mx, 4));
      mx = fmaxf(mx, __shfl_xor(mx, 8));
      float nm = fmaxf(runm[r], mx);
      float al = __expf(runm[r] - nm);
      float p0 = __expf(s0[r] - nm);
      float p1 = __expf(s1[r] - nm);
      float ts = p0 + p1;
      ts += __shfl_xor(ts, 1);
      ts += __shfl_xor(ts, 2);
      ts += __shfl_xor(ts, 4);
      ts += __shfl_xor(ts, 8);
      runs[r] = runs[r] * al + ts;
      runm[r] = nm;
      alpha[r] = al;
      s0[r] = p0; s1[r] = p1;
    }
#pragma unroll
    for (int ni = 0; ni < 4; ++ni)
#pragma unroll
      for (int r = 0; r < 8; ++r) o_acc[ni][r] *= alpha[r];

    // restage P (C layout) into LDS, reread as A-fragment (per-wave region)
#pragma unroll
    for (int r = 0; r < 8; ++r) {
      myP[(r + 8 * half) * 32 + lr]      = (_Float16)s0[r];
      myP[(r + 8 * half) * 32 + 16 + lr] = (_Float16)s1[r];
    }
    v16h pf = ld_fragA(myP + lr * 32, half);
#pragma unroll
    for (int ni = 0; ni < 4; ++ni) {
      v16h vf = ld_fragB(vt + (size_t)(ni * 16 + lr) * TB_T + kbase, half);
      o_acc[ni] = wmma_f16(pf, vf, o_acc[ni]);
    }
  }

#pragma unroll
  for (int ni = 0; ni < 4; ++ni)
#pragma unroll
    for (int r = 0; r < 8; ++r) {
      int tok = q0 + r + 8 * half;
      float val = o_acc[ni][r] / runs[r];
      O[((size_t)(b * TB_T + tok)) * TB_C + h * TB_D + ni * 16 + lr] = (_Float16)val;
    }
}

// ---------------------------------------------------------------------------
// Router: logits = h2 @ w_router; top-2 softmax -> dense weight vector [M][E]
// ---------------------------------------------------------------------------
__global__ void tb_router(const _Float16* __restrict__ h2, const float* __restrict__ wr,
                          float* __restrict__ wts) {
  int t = blockIdx.x * blockDim.x + threadIdx.x;
  if (t >= TB_M) return;
  const _Float16* hr = h2 + (size_t)t * TB_C;
  float l[4] = {0.f, 0.f, 0.f, 0.f};
  for (int c = 0; c < TB_C; ++c) {
    float hv = (float)hr[c];
    float4 w = ((const float4*)wr)[c];
    l[0] += hv * w.x; l[1] += hv * w.y; l[2] += hv * w.z; l[3] += hv * w.w;
  }
  int i1 = 0;
  for (int e = 1; e < 4; ++e) if (l[e] > l[i1]) i1 = e;
  int i2 = -1;
  for (int e = 0; e < 4; ++e) { if (e == i1) continue; if (i2 < 0 || l[e] > l[i2]) i2 = e; }
  float e2 = __expf(l[i2] - l[i1]);
  float w1 = 1.f / (1.f + e2), w2 = e2 / (1.f + e2);
  float o[4] = {0.f, 0.f, 0.f, 0.f};
  o[i1] = w1; o[i2] = w2;
  wts[t * 4 + 0] = o[0]; wts[t * 4 + 1] = o[1];
  wts[t * 4 + 2] = o[2]; wts[t * 4 + 3] = o[3];
}

__global__ void tb_add(const float* __restrict__ a, const float* __restrict__ b,
                       float* __restrict__ o, int n) {
  int i = blockIdx.x * blockDim.x + threadIdx.x;
  if (i < n) o[i] = a[i] + b[i];
}

// ---------------------------------------------------------------------------
extern "C" void kernel_launch(void* const* d_in, const int* in_sizes, int n_in,
                              void* d_out, int out_size, void* d_ws, size_t ws_size,
                              hipStream_t stream) {
  (void)in_sizes; (void)n_in; (void)out_size; (void)ws_size;
  const float* x    = (const float*)d_in[0];
  const float* g1   = (const float*)d_in[1];
  const float* g2   = (const float*)d_in[2];
  const float* wqkv = (const float*)d_in[3];
  const float* bqkv = (const float*)d_in[4];
  const float* wo   = (const float*)d_in[5];
  const float* bo   = (const float*)d_in[6];
  const float* wr   = (const float*)d_in[7];
  const float* wg   = (const float*)d_in[8];
  const float* bg   = (const float*)d_in[9];
  const float* wu   = (const float*)d_in[10];
  const float* bu   = (const float*)d_in[11];
  const float* wd   = (const float*)d_in[12];
  const float* bd   = (const float*)d_in[13];
  float* out = (float*)d_out;

  size_t off = 0;
  auto alloc = [&](size_t bytes) -> void* {
    void* p = (char*)d_ws + off;
    off = (off + bytes + 255) & ~(size_t)255;
    return p;
  };
  _Float16* h1    = (_Float16*)alloc((size_t)TB_M * TB_C * 2);
  _Float16* wqkvT = (_Float16*)alloc((size_t)TB_3C * TB_C * 2);
  _Float16* qkv   = (_Float16*)alloc((size_t)TB_M * TB_3C * 2);   // reused as `gate`
  _Float16* Qb    = (_Float16*)alloc((size_t)TB_B * TB_H * TB_T * TB_D * 2); // reused as `act`
  _Float16* Kb    = (_Float16*)alloc((size_t)TB_B * TB_H * TB_T * TB_D * 2);
  _Float16* VTb   = (_Float16*)alloc((size_t)TB_B * TB_H * TB_T * TB_D * 2);
  _Float16* Obuf  = (_Float16*)alloc((size_t)TB_M * TB_C * 2);
  _Float16* woT   = (_Float16*)alloc((size_t)TB_C * TB_C * 2);
  float*    x1    = (float*)alloc((size_t)TB_M * TB_C * 4);
  _Float16* h2    = (_Float16*)alloc((size_t)TB_M * TB_C * 2);
  float*    wts   = (float*)alloc((size_t)TB_M * TB_E * 4);
  _Float16* wgT   = (_Float16*)alloc((size_t)TB_E * TB_HIDP * TB_C * 2);
  _Float16* wuT   = (_Float16*)alloc((size_t)TB_E * TB_HIDP * TB_C * 2);
  _Float16* wdT   = (_Float16*)alloc((size_t)TB_E * TB_C * TB_HIDP * 2);
  float*    mixed = (float*)alloc((size_t)TB_M * TB_C * 4);
  _Float16* gate  = qkv;   // qkv dead after RoPE; 24 MB >= 23.1 MB
  _Float16* act   = Qb;    // Q/K/VT dead after attention; contiguous 24 MB
  (void)Kb; (void)VTb;

  dim3 tb(32, 8);
  // --- weight conversion/transposition to f16 (HID padded 2730 -> 2816) ---
  tb_transpose_f16<<<dim3(TB_C / 32, TB_3C / 32), tb, 0, stream>>>(
      wqkv, TB_C, TB_3C, wqkvT, TB_3C, TB_C);
  tb_transpose_f16<<<dim3(TB_C / 32, TB_C / 32), tb, 0, stream>>>(
      wo, TB_C, TB_C, woT, TB_C, TB_C);
  for (int e = 0; e < TB_E; ++e) {
    tb_transpose_f16<<<dim3(TB_C / 32, TB_HIDP / 32), tb, 0, stream>>>(
        wg + (size_t)e * TB_C * TB_HID, TB_C, TB_HID,
        wgT + (size_t)e * TB_HIDP * TB_C, TB_HIDP, TB_C);
    tb_transpose_f16<<<dim3(TB_C / 32, TB_HIDP / 32), tb, 0, stream>>>(
        wu + (size_t)e * TB_C * TB_HID, TB_C, TB_HID,
        wuT + (size_t)e * TB_HIDP * TB_C, TB_HIDP, TB_C);
    tb_transpose_f16<<<dim3(TB_HIDP / 32, TB_C / 32), tb, 0, stream>>>(
        wd + (size_t)e * TB_HID * TB_C, TB_HID, TB_C,
        wdT + (size_t)e * TB_C * TB_HIDP, TB_C, TB_HIDP);
  }

  // --- attention path ---
  tb_rmsnorm<<<TB_M, 256, 0, stream>>>(x, g1, h1);
  tb_gemm<0><<<dim3(TB_M / 128, TB_3C / 128), 256, 0, stream>>>(
      h1, wqkvT, bqkv, TB_M, TB_3C, TB_C, TB_3C,
      qkv, nullptr, nullptr, nullptr, nullptr, 0, nullptr);
  tb_rope<<<TB_M, 256, 0, stream>>>(qkv, Qb, Kb, VTb);
  tb_attn<<<dim3(TB_T / 64, TB_B * TB_H), 128, 0, stream>>>(Qb, Kb, VTb, Obuf);
  tb_gemm<1><<<dim3(TB_M / 128, TB_C / 128), 256, 0, stream>>>(
      Obuf, woT, bo, TB_M, TB_C, TB_C, TB_C,
      nullptr, x1, x, nullptr, nullptr, 0, nullptr);

  // --- MoE path ---
  tb_rmsnorm<<<TB_M, 256, 0, stream>>>(x1, g2, h2);
  tb_router<<<TB_M / 256, 256, 0, stream>>>(h2, wr, wts);
  for (int e = 0; e < TB_E; ++e) {
    tb_gemm<2><<<dim3(TB_M / 128, TB_HIDP / 128), 256, 0, stream>>>(
        h2, wgT + (size_t)e * TB_HIDP * TB_C, bg + (size_t)e * TB_HID,
        TB_M, TB_HIDP, TB_C, TB_HID,
        gate, nullptr, nullptr, nullptr, nullptr, 0, nullptr);
    tb_gemm<3><<<dim3(TB_M / 128, TB_HIDP / 128), 256, 0, stream>>>(
        h2, wuT + (size_t)e * TB_HIDP * TB_C, bu + (size_t)e * TB_HID,
        TB_M, TB_HIDP, TB_C, TB_HID,
        act, nullptr, nullptr, gate, nullptr, 0, nullptr);
    tb_gemm<4><<<dim3(TB_M / 128, TB_C / 128), 256, 0, stream>>>(
        act, wdT + (size_t)e * TB_C * TB_HIDP, bd + (size_t)e * TB_C,
        TB_M, TB_C, TB_HIDP, TB_C,
        nullptr, nullptr, nullptr, nullptr, wts, e, mixed);
  }
  tb_add<<<(TB_M * TB_C) / 256, 256, 0, stream>>>(x1, mixed, out, TB_M * TB_C);
}
